// TabNetRegressor_48567490183738
// MI455X (gfx1250) — matrix operand order repaired and checked by reading, not compiled, not run on hardware
//
#include <hip/hip_runtime.h>
#include <hip/hip_bf16.h>

// ---------------------------------------------------------------------------
// TabNet forward, fully fused per 128-row virtual batch (ghost-BN block).
// MI455X / gfx1250: wave32, WMMA bf16 (16x16x32, f32 accum), 258KB LDS/WG.
// Dual N-tile accumulation: two independent WMMA chains per wave to hide
// XDL latency; A fragments shared across the pair (halves LDS reads).
// ---------------------------------------------------------------------------

typedef __bf16 bf16_t;
typedef __attribute__((ext_vector_type(16))) __bf16 v16bf;
typedef __attribute__((ext_vector_type(8)))  __bf16 v8bf;
typedef __attribute__((ext_vector_type(8)))  float  v8f;

#define TB_B      65536
#define TB_D      256
#define TB_NDA    128
#define TB_NA     64
#define TB_ND     64
#define TB_VBS    128
#define TB_NSTEPS 5
#define TB_EPS    1e-5f
#define TB_GAMMA  1.3f
#define TB_SCALE  0.70710678118654752f

// ---------------------------------------------------------------------------
// Pre-pass: cast f32 weights (mat, K, N) -> bf16 transposed (mat, N, K)
// so WMMA B-fragments are contiguous-K per lane (lane = N column).
// ---------------------------------------------------------------------------
__global__ void tabnet_wcast_kernel(const float* __restrict__ src,
                                    bf16_t* __restrict__ dst,
                                    int K, int N, long total) {
    long i = (long)blockIdx.x * blockDim.x + threadIdx.x;
    if (i >= total) return;
    long per = (long)K * N;
    long mat = i / per;
    long rem = i - mat * per;
    int  n   = (int)(rem / K);
    int  k   = (int)(rem - (long)n * K);
    dst[i] = (bf16_t)src[mat * per + (long)k * N + n];
}

// ---------------------------------------------------------------------------
// WMMA matmul: C(128 x 256, bf16, ldc=256 in LDS) = A(128 x K, bf16) @ W^T
// Wt is (256 x K) row-major bf16. Wave w owns M-tile w; N-tiles in pairs.
// ---------------------------------------------------------------------------
template <int K, int LDA>
__device__ __forceinline__ void mm_wmma(const bf16_t* __restrict__ A,
                                        const bf16_t* __restrict__ Wt,
                                        bf16_t* __restrict__ C,
                                        int wave, int lane) {
    const int lrow = lane & 15;   // M index (A) / N index (B) within tile
    const int lhi  = lane >> 4;   // selects K half (contiguous k = 16*lhi + s, not ISA table order)
    const int koff = lhi * 16;    // A: K = {koff..koff+15}
    const int mtile = wave;

    const bf16_t* arow = A + (mtile * 16 + lrow) * LDA;

    for (int nt = 0; nt < 16; nt += 2) {
        v8f acc0 = {};
        v8f acc1 = {};
        const bf16_t* brow0 = Wt + (nt * 16 + lrow) * K + lhi * 16; // 16 contiguous K
        const bf16_t* brow1 = brow0 + 16 * K;
#pragma unroll
        for (int ks = 0; ks < K / 32; ++ks) {
            const bf16_t* pa = arow + ks * 32 + koff;
            v8bf alo = *(const v8bf*)pa;
            v8bf ahi = *(const v8bf*)(pa + 8);
            v16bf a = __builtin_shufflevector(alo, ahi,
                0, 1, 2, 3, 4, 5, 6, 7, 8, 9, 10, 11, 12, 13, 14, 15);
            v16bf b0 = *(const v16bf*)(brow0 + ks * 32);
            v16bf b1 = *(const v16bf*)(brow1 + ks * 32);
            acc0 = __builtin_amdgcn_wmma_f32_16x16x32_bf16(
                false, a, false, b0, (short)0, acc0, false, false);
            acc1 = __builtin_amdgcn_wmma_f32_16x16x32_bf16(
                false, a, false, b1, (short)0, acc1, false, false);
        }
        // C/D layout: lane<16 -> (M=v, N=lrow); lane>=16 -> (M=v+8, N=lrow)
        const int n0    = nt * 16 + lrow;
        const int mbase = mtile * 16 + 8 * lhi;
#pragma unroll
        for (int v = 0; v < 8; ++v) {
            C[(mbase + v) * TB_D + n0]      = (bf16_t)acc0[v];
            C[(mbase + v) * TB_D + n0 + 16] = (bf16_t)acc1[v];
        }
    }
}

// Per-column ghost-BN stats over the 128-row block; folds gamma/beta into
// (scale, shift). One column per thread -> conflict-free LDS.
__device__ __forceinline__ void bn_stats(const bf16_t* __restrict__ H,
                                         const float* __restrict__ g,
                                         const float* __restrict__ b,
                                         float* __restrict__ s_sc,
                                         float* __restrict__ s_sh, int tid) {
    float s = 0.f, ss = 0.f;
#pragma unroll 4
    for (int r = 0; r < TB_VBS; ++r) {
        float v = (float)H[r * TB_D + tid];
        s += v; ss += v * v;
    }
    float m    = s * (1.f / TB_VBS);
    float var  = ss * (1.f / TB_VBS) - m * m;
    float rstd = rsqrtf(var + TB_EPS);
    float sc   = rstd * g[tid];
    s_sc[tid]  = sc;
    s_sh[tid]  = b[tid] - m * sc;
}

__device__ __forceinline__ void prefetch_bytes(const void* p, int bytes, int tid) {
    const char* c = (const char*)p;
    for (int off = tid * 128; off < bytes; off += 256 * 128)
        __builtin_prefetch(c + off, 0, 0);
}

// ---------------------------------------------------------------------------
// Main fused kernel: grid = 512 blocks (one per 128-row virtual batch),
// block = 256 threads = 8 wave32.
// ---------------------------------------------------------------------------
__global__ __launch_bounds__(256) void tabnet_fused_kernel(
    const float* __restrict__ x, const float* __restrict__ g0,
    const float* __restrict__ b0,
    const bf16_t* __restrict__ ws0t,                       // 5 x 256 x 256
    const float* __restrict__ g_s0, const float* __restrict__ b_s0,
    const bf16_t* __restrict__ wglut,                      // 15 x 256 x 128
    const float* __restrict__ g_glu, const float* __restrict__ b_glu,
    const bf16_t* __restrict__ watt_t,                     // 5 x 256 x 64
    const float* __restrict__ g_att, const float* __restrict__ b_att,
    const float* __restrict__ wfinal,                      // 64
    float* __restrict__ out) {

    extern __shared__ __align__(16) char smem_raw[];
    bf16_t* xb   = (bf16_t*)smem_raw;           // 128x256 bf16  (x, live)
    bf16_t* pri  = xb + TB_VBS * TB_D;          // 128x256 bf16  (prior)
    bf16_t* hpre = pri + TB_VBS * TB_D;         // 128x256 bf16  (pre-BN scratch)
    bf16_t* hglu = hpre + TB_VBS * TB_D;        // 128x128 bf16  (post-GLU h)
    float*  resb = (float*)(hglu + TB_VBS * TB_NDA); // 128x64 f32
    float*  s_sc = resb + TB_VBS * TB_ND;       // 256 f32
    float*  s_sh = s_sc + TB_D;                 // 256 f32

    const int tid  = threadIdx.x;
    const int wave = tid >> 5;
    const int lane = tid & 31;
    const int blk  = blockIdx.x;

    const float* xg = x + (long)blk * TB_VBS * TB_D;

    // ---- phase 0: ghost_bn(x, g0, b0) -> xb (bf16); init prior, res ----
    {
        const int c = tid;
        float s = 0.f, ss = 0.f;
#pragma unroll 4
        for (int r = 0; r < TB_VBS; ++r) {
            float v = xg[r * TB_D + c];
            s += v; ss += v * v;
        }
        float m    = s * (1.f / TB_VBS);
        float rstd = rsqrtf(ss * (1.f / TB_VBS) - m * m + TB_EPS);
        float sc = rstd * g0[c];
        float sh = b0[c] - m * sc;
        for (int r = 0; r < TB_VBS; ++r) {
            xb[r * TB_D + c]  = (bf16_t)(xg[r * TB_D + c] * sc + sh);
            pri[r * TB_D + c] = (bf16_t)1.0f;
        }
        for (int i = tid; i < TB_VBS * TB_ND; i += 256) resb[i] = 0.f;
    }
    __syncthreads();

    for (int s = 0; s < TB_NSTEPS; ++s) {
        const bf16_t* Ws0s = ws0t + (long)s * TB_D * TB_D;
        prefetch_bytes(Ws0s, TB_D * TB_D * 2, tid);
        prefetch_bytes(wglut + (long)(s * 3) * TB_D * TB_NDA,
                       3 * TB_D * TB_NDA * 2, tid);

        // ---- h0 = ghost_bn(x @ W_s0[s]) -> GLU -> hglu ----
        mm_wmma<TB_D, TB_D>(xb, Ws0s, hpre, wave, lane);
        __syncthreads();
        bn_stats(hpre, g_s0 + s * TB_D, b_s0 + s * TB_D, s_sc, s_sh, tid);
        __syncthreads();
        {   // GLU: out * sigmoid(gate), replace hglu
            const int c  = tid & (TB_NDA - 1);
            const int r0 = (tid >> 7) * 64;
            const float scA = s_sc[c],            shA = s_sh[c];
            const float scB = s_sc[c + TB_NDA],   shB = s_sh[c + TB_NDA];
            for (int r = r0; r < r0 + 64; ++r) {
                float o  = (float)hpre[r * TB_D + c] * scA + shA;
                float gt = (float)hpre[r * TB_D + c + TB_NDA] * scB + shB;
                hglu[r * TB_NDA + c] = (bf16_t)(o * (1.f / (1.f + __expf(-gt))));
            }
        }
        __syncthreads();

        // ---- three GLU blocks on W_glu[s, j] (j=0 replace, j=1,2 residual) ----
        for (int j = 0; j < 3; ++j) {
            const bf16_t* Wg = wglut + (long)(s * 3 + j) * TB_D * TB_NDA;
            mm_wmma<TB_NDA, TB_NDA>(hglu, Wg, hpre, wave, lane);
            __syncthreads();
            bn_stats(hpre, g_glu + (s * 3 + j) * TB_D, b_glu + (s * 3 + j) * TB_D,
                     s_sc, s_sh, tid);
            __syncthreads();
            {
                const int c  = tid & (TB_NDA - 1);
                const int r0 = (tid >> 7) * 64;
                const float scA = s_sc[c],          shA = s_sh[c];
                const float scB = s_sc[c + TB_NDA], shB = s_sh[c + TB_NDA];
                for (int r = r0; r < r0 + 64; ++r) {
                    float o   = (float)hpre[r * TB_D + c] * scA + shA;
                    float gt  = (float)hpre[r * TB_D + c + TB_NDA] * scB + shB;
                    float val = o * (1.f / (1.f + __expf(-gt)));
                    if (j > 0) val += TB_SCALE * (float)hglu[r * TB_NDA + c];
                    hglu[r * TB_NDA + c] = (bf16_t)val;
                }
            }
            __syncthreads();
        }

        // ---- res += relu(h[:, :64]) ----
        {
            const int c  = tid & (TB_ND - 1);
            const int r0 = (tid >> 6) * 32;
            for (int r = r0; r < r0 + 32; ++r) {
                float dv = (float)hglu[r * TB_NDA + c];
                resb[r * TB_ND + c] += fmaxf(dv, 0.f);
            }
        }

        // ---- att = ghost_bn(h[:, 64:] @ W_att[s]) ----
        mm_wmma<TB_NA, TB_NDA>(hglu + TB_NA, watt_t + (long)s * TB_D * TB_NA,
                               hpre, wave, lane);
        __syncthreads();
        bn_stats(hpre, g_att + s * TB_D, b_att + s * TB_D, s_sc, s_sh, tid);
        __syncthreads();

        // ---- M = softmax(prior*att); x *= M; prior *= (GAMMA - M) ----
        if (tid < TB_VBS) {
            const int r = tid;
            bf16_t* hrow = hpre + r * TB_D;
            float mx = -3.0e38f;
            for (int cc = 0; cc < TB_D; ++cc) {        // skewed: conflict-free
                int c = (cc + r) & (TB_D - 1);
                float att = (float)hrow[c] * s_sc[c] + s_sh[c];
                float z   = (float)pri[r * TB_D + c] * att;
                hrow[c] = (bf16_t)z;
                mx = fmaxf(mx, z);
            }
            float sum = 0.f;
            for (int cc = 0; cc < TB_D; ++cc) {
                int c = (cc + r) & (TB_D - 1);
                float e = __expf((float)hrow[c] - mx);
                sum += e;
                hrow[c] = (bf16_t)e;
            }
            float inv = 1.f / sum;
            for (int cc = 0; cc < TB_D; ++cc) {
                int c = (cc + r) & (TB_D - 1);
                float M  = (float)hrow[c] * inv;
                float xv = (float)xb[r * TB_D + c];
                float pv = (float)pri[r * TB_D + c];
                xb[r * TB_D + c]  = (bf16_t)(M * xv);
                pri[r * TB_D + c] = (bf16_t)((TB_GAMMA - M) * pv);
            }
        }
        __syncthreads();
    }

    // ---- out = res @ W_final (64 x 1) ----
    if (tid < TB_VBS) {
        const int r = tid;
        float acc = 0.f;
#pragma unroll
        for (int k = 0; k < TB_ND; ++k)
            acc += resb[r * TB_ND + k] * wfinal[k];
        out[(long)blk * TB_VBS + r] = acc;
    }
}

// ---------------------------------------------------------------------------
extern "C" void kernel_launch(void* const* d_in, const int* in_sizes, int n_in,
                              void* d_out, int out_size, void* d_ws, size_t ws_size,
                              hipStream_t stream) {
    (void)in_sizes; (void)n_in; (void)out_size; (void)ws_size;

    const float* x_in   = (const float*)d_in[0];
    const float* g0     = (const float*)d_in[1];
    const float* b0     = (const float*)d_in[2];
    const float* Ws0    = (const float*)d_in[3];   // (5, 256, 256)
    const float* g_s0   = (const float*)d_in[4];
    const float* b_s0   = (const float*)d_in[5];
    const float* Wglu   = (const float*)d_in[6];   // (5, 3, 128, 256)
    const float* g_glu  = (const float*)d_in[7];
    const float* b_glu  = (const float*)d_in[8];
    const float* Watt   = (const float*)d_in[9];   // (5, 64, 256)
    const float* g_att  = (const float*)d_in[10];
    const float* b_att  = (const float*)d_in[11];
    const float* Wf     = (const float*)d_in[12];  // (64, 1)
    float* out          = (float*)d_out;

    // d_ws layout: bf16 transposed weights (N x K per matrix)
    bf16_t* ws0t   = (bf16_t*)d_ws;                       // 5*256*256
    bf16_t* wglut  = ws0t  + 5L * 256 * 256;              // 15*256*128
    bf16_t* watt_t = wglut + 15L * 256 * 128;             // 5*256*64

    {
        long tot = 5L * 256 * 256;
        tabnet_wcast_kernel<<<(tot + 255) / 256, 256, 0, stream>>>(Ws0, ws0t, 256, 256, tot);
    }
    {
        long tot = 15L * 128 * 256;
        tabnet_wcast_kernel<<<(tot + 255) / 256, 256, 0, stream>>>(Wglu, wglut, 128, 256, tot);
    }
    {
        long tot = 5L * 64 * 256;
        tabnet_wcast_kernel<<<(tot + 255) / 256, 256, 0, stream>>>(Watt, watt_t, 64, 256, tot);
    }

    // LDS: 3*(128*256*2) + 128*128*2 + 128*64*4 + 2*256*4 = 264192 bytes
    const size_t smem =
        (size_t)(3 * TB_VBS * TB_D + TB_VBS * TB_NDA) * sizeof(bf16_t) +
        (size_t)(TB_VBS * TB_ND + 2 * TB_D) * sizeof(float);
    hipFuncSetAttribute((const void*)tabnet_fused_kernel,
                        hipFuncAttributeMaxDynamicSharedMemorySize, (int)smem);

    tabnet_fused_kernel<<<TB_B / TB_VBS, 256, smem, stream>>>(
        x_in, g0, b0, ws0t, g_s0, b_s0, wglut, g_glu, b_glu,
        watt_t, g_att, b_att, Wf, out);
}
